// TransfomerAttentionLayer_44332652430161
// MI455X (gfx1250) — matrix operand order repaired and checked by reading, not compiled, not run on hardware
//
#include <hip/hip_runtime.h>
#include <hip/hip_bf16.h>

// Problem constants (match reference)
#define N_DST_C 25000
#define E_C     400000
#define DM      128      // DN = DE = DT = DO = 128
#define HEADS   8
#define LN_EPS  1e-5f

typedef __attribute__((ext_vector_type(16))) __bf16 v16bf;
typedef __attribute__((ext_vector_type(8)))  float  v8f;

static __device__ __forceinline__ v8f wmma_bf16(v16bf a, v16bf b, v8f c) {
  // D = A(16x32 bf16) * B(32x16 bf16) + C(16x16 f32)
  return __builtin_amdgcn_wmma_f32_16x16x32_bf16(false, a, false, b, (short)0, c,
                                                 false, false);
}

// Monotone order-preserving float <-> uint encoding for atomicMax on f32 scores.
static __device__ __forceinline__ unsigned fenc(float f) {
  unsigned u = __float_as_uint(f);
  return (u & 0x80000000u) ? ~u : (u | 0x80000000u);
}
static __device__ __forceinline__ float fdec(unsigned k) {
  unsigned u = (k & 0x80000000u) ? (k & 0x7fffffffu) : ~k;
  return __uint_as_float(u);
}

// ---------------------------------------------------------------- utilities
__global__ void k_zero(float* __restrict__ p, int n) {
  int i = blockIdx.x * 256 + threadIdx.x;
  if (i < n) p[i] = 0.0f;
}

// Convert a row-major fp32 weight W[128 x rowStride] into bf16 B-fragments.
// Fragment f = ks*8 + nt covers B[k][n] with n = nt*16+(lane&15),
// k = ks*32 + (i<8 ? i : i+8) + 8*(lane>>4); stored as out[f*512 + lane*16 + i].
__global__ void k_prep_w(const float* __restrict__ W, int rowStride,
                         __bf16* __restrict__ out, int total) {
  int idx = blockIdx.x * 256 + threadIdx.x;
  if (idx >= total) return;
  int f      = idx >> 9;
  int within = idx & 511;
  int lane   = within >> 4;
  int i      = within & 15;
  int nt = f & 7, ks = f >> 3;
  int n = nt * 16 + (lane & 15);
  int k = ks * 32 + (i < 8 ? i : i + 8) + 8 * (lane >> 4);
  out[idx] = (__bf16)W[(size_t)n * rowStride + k];
}

// bq_eff[o] = bq[o] + sum_t Wq[o][128+t] * cos(tb[t])
__global__ void k_prep_bq(const float* __restrict__ Wq, const float* __restrict__ bq,
                          const float* __restrict__ tb, float* __restrict__ bq_eff) {
  int o = threadIdx.x;
  if (o >= DM) return;
  float s = bq[o];
  for (int t = 0; t < DM; ++t) s += Wq[(size_t)o * 256 + 128 + t] * __cosf(tb[t]);
  bq_eff[o] = s;
}

// ------------------------------------------------- generic A-fragment pieces
// A-matrix 16x32 bf16 layout: lane holds row m = lane&15.
// i in [0,16): k = ks*32 + (i<8 ? i : i+8) + 8*(lane>>4)
// => two contiguous 8-float runs at k0 = ks*32 + 8*half (+16).
static __device__ __forceinline__ void fill8(float* v, const float* __restrict__ p) {
  float4 x = *(const float4*)p;
  float4 y = *(const float4*)(p + 4);
  v[0] = x.x; v[1] = x.y; v[2] = x.z; v[3] = x.w;
  v[4] = y.x; v[5] = y.y; v[6] = y.z; v[7] = y.w;
}
static __device__ __forceinline__ v16bf to_frag(const float* v) {
  v16bf a;
#pragma unroll
  for (int i = 0; i < 16; ++i) a[i] = (__bf16)v[i];
  return a;
}

// GEMM for the kv_in rows: [source_h(128) | edge_feats(128) | cos(dt*tw+tb)(128)]
static __device__ __forceinline__ void kv_gemm(const float* __restrict__ srow,
                                               const float* __restrict__ frow,
                                               float dte,
                                               const float* __restrict__ tw,
                                               const float* __restrict__ tb,
                                               const __bf16* __restrict__ WB,
                                               int lane, v8f acc[8]) {
  const int half = lane >> 4;
#pragma unroll
  for (int ks = 0; ks < 12; ++ks) {
    const int seg = ks >> 2;
    float vals[16];
#pragma unroll
    for (int run = 0; run < 2; ++run) {
      const int off = (ks & 3) * 32 + 8 * half + 16 * run;
      if (seg == 0) {
        fill8(vals + run * 8, srow + off);
      } else if (seg == 1) {
        fill8(vals + run * 8, frow + off);
      } else {
        float w[8], b[8];
        fill8(w, tw + off);
        fill8(b, tb + off);
#pragma unroll
        for (int j = 0; j < 8; ++j) vals[run * 8 + j] = __cosf(fmaf(dte, w[j], b[j]));
      }
    }
    v16bf a = to_frag(vals);
#pragma unroll
    for (int nt = 0; nt < 8; ++nt) {
      v16bf b = *(const v16bf*)(WB + (((ks << 3) + nt) << 9) + (lane << 4));
      acc[nt] = wmma_bf16(a, b, acc[nt]);
    }
  }
}

// GEMM for node rows built from up to two 128-wide fp32 segments.
template <int NKS>
static __device__ __forceinline__ void rows2_gemm(const float* __restrict__ r0,
                                                  const float* __restrict__ r1,
                                                  const __bf16* __restrict__ WB,
                                                  int lane, v8f acc[8]) {
  const int half = lane >> 4;
#pragma unroll
  for (int ks = 0; ks < NKS; ++ks) {
    const float* r = (ks < 4) ? r0 : r1;
    float vals[16];
#pragma unroll
    for (int run = 0; run < 2; ++run) {
      const int off = (ks & 3) * 32 + 8 * half + 16 * run;
      fill8(vals + run * 8, r + off);
    }
    v16bf a = to_frag(vals);
#pragma unroll
    for (int nt = 0; nt < 8; ++nt) {
      v16bf b = *(const v16bf*)(WB + (((ks << 3) + nt) << 9) + (lane << 4));
      acc[nt] = wmma_bf16(a, b, acc[nt]);
    }
  }
}

// -------------------------------------------------------------- Qn = proj(q)
__global__ __launch_bounds__(256) void k_qn(const float* __restrict__ target_h,
                                            const __bf16* __restrict__ WqB,
                                            const float* __restrict__ bq_eff,
                                            float* __restrict__ Qn) {
  const int wv = threadIdx.x >> 5, lane = threadIdx.x & 31;
  const int half = lane >> 4, nlo = lane & 15;
  const int row0 = (blockIdx.x * 8 + wv) * 16;
  if (row0 >= N_DST_C) return;  // no barriers in this kernel

  const int arow = min(row0 + nlo, N_DST_C - 1);
  const float* r0 = target_h + (size_t)arow * DM;

  v8f acc[8] = {};
  rows2_gemm<4>(r0, r0, WqB, lane, acc);

#pragma unroll
  for (int nt = 0; nt < 8; ++nt) {
    const float bias = bq_eff[nt * 16 + nlo];
#pragma unroll
    for (int r = 0; r < 8; ++r) {
      const int row = row0 + 8 * half + r;
      if (row < N_DST_C) Qn[(size_t)row * DM + nt * 16 + nlo] = acc[nt][r] + bias;
    }
  }
}

// ------------------------------------------- pass 1: K, scores, segment max
__global__ __launch_bounds__(256) void k_edge_score(
    const float* __restrict__ source_h, const float* __restrict__ edge_feats,
    const float* __restrict__ delta_time, const int* __restrict__ edge_dst,
    const float* __restrict__ tw, const float* __restrict__ tb,
    const float* __restrict__ bk, const __bf16* __restrict__ WkB,
    const float* __restrict__ Qn, float* __restrict__ scorebuf,
    unsigned* __restrict__ mbuf) {
  __shared__ float kbuf[8][16][DM];  // 64 KB
  const int wv = threadIdx.x >> 5, lane = threadIdx.x & 31;
  const int half = lane >> 4, nlo = lane & 15;
  const int e0 = blockIdx.x * 128 + wv * 16;

  const int erow = e0 + nlo;
  const float* srow = source_h + (size_t)erow * DM;
  const float* frow = edge_feats + (size_t)erow * DM;
  const float dte = delta_time[erow];

  v8f acc[8] = {};
  kv_gemm(srow, frow, dte, tw, tb, WkB, lane, acc);

#pragma unroll
  for (int nt = 0; nt < 8; ++nt) {
    const float bias = bk[nt * 16 + nlo];
#pragma unroll
    for (int r = 0; r < 8; ++r)
      kbuf[wv][8 * half + r][nt * 16 + nlo] = acc[nt][r] + bias;
  }
  __syncthreads();

#pragma unroll
  for (int t = 0; t < 4; ++t) {
    const int task = t * 32 + lane;          // 16 edges x 8 heads
    const int el = task >> 3, h = task & 7;
    const int e = e0 + el;
    const int d = edge_dst[e];
    const float* q = Qn + (size_t)d * DM + h * 16;
    float s = 0.0f;
#pragma unroll
    for (int j = 0; j < 16; ++j) s = fmaf(kbuf[wv][el][h * 16 + j], q[j], s);
    s = (s > 0.0f) ? s : 0.2f * s;           // leaky_relu(0.2)
    scorebuf[(size_t)e * HEADS + h] = s;
    atomicMax(mbuf + (size_t)d * HEADS + h, fenc(s));
  }
}

// ------------------------------------------------ pass 2: exp + denominator
__global__ __launch_bounds__(256) void k_soft(const float* __restrict__ scorebuf,
                                              const int* __restrict__ edge_dst,
                                              const unsigned* __restrict__ mbuf,
                                              float* __restrict__ exbuf,
                                              float* __restrict__ den) {
  const int i = blockIdx.x * 256 + threadIdx.x;
  if (i >= E_C * HEADS) return;
  const int e = i >> 3, h = i & 7;
  const int d = edge_dst[e];
  const float mm = fdec(mbuf[(size_t)d * HEADS + h]);
  const float ex = __expf(scorebuf[i] - mm);
  exbuf[i] = ex;
  unsafeAtomicAdd(den + (size_t)d * HEADS + h, ex);
}

// -------------------------------- pass 3: V, attention weight, scatter-add
__global__ __launch_bounds__(256) void k_edge_agg(
    const float* __restrict__ source_h, const float* __restrict__ edge_feats,
    const float* __restrict__ delta_time, const int* __restrict__ edge_dst,
    const float* __restrict__ tw, const float* __restrict__ tb,
    const float* __restrict__ bv, const __bf16* __restrict__ WvB,
    const float* __restrict__ exbuf, const float* __restrict__ den,
    float* __restrict__ agg) {
  __shared__ float attbuf[8][16][HEADS];
  __shared__ int dstbuf[8][16];
  const int wv = threadIdx.x >> 5, lane = threadIdx.x & 31;
  const int half = lane >> 4, nlo = lane & 15;
  const int e0 = blockIdx.x * 128 + wv * 16;

  const int erow = e0 + nlo;
  const float* srow = source_h + (size_t)erow * DM;
  const float* frow = edge_feats + (size_t)erow * DM;
  const float dte = delta_time[erow];

  v8f acc[8] = {};
  kv_gemm(srow, frow, dte, tw, tb, WvB, lane, acc);

#pragma unroll
  for (int t = 0; t < 4; ++t) {
    const int task = t * 32 + lane;
    const int el = task >> 3, h = task & 7;
    const int e = e0 + el;
    const int d = edge_dst[e];
    attbuf[wv][el][h] = exbuf[(size_t)e * HEADS + h] / den[(size_t)d * HEADS + h];
  }
  if (lane < 16) dstbuf[wv][lane] = edge_dst[e0 + lane];
  __syncthreads();

#pragma unroll
  for (int nt = 0; nt < 8; ++nt) {
    const float bias = bv[nt * 16 + nlo];
#pragma unroll
    for (int r = 0; r < 8; ++r) {
      const int m = 8 * half + r;
      const float v = (acc[nt][r] + bias) * attbuf[wv][m][nt];
      unsafeAtomicAdd(agg + (size_t)dstbuf[wv][m] * DM + nt * 16 + nlo, v);
    }
  }
}

// ------------------------- output: GEMM + ReLU + LayerNorm, fused per tile
__global__ __launch_bounds__(256) void k_out(const float* __restrict__ agg,
                                             const float* __restrict__ target_h,
                                             const __bf16* __restrict__ WoB,
                                             const float* __restrict__ bo,
                                             const float* __restrict__ ln_g,
                                             const float* __restrict__ ln_b,
                                             float* __restrict__ out) {
  __shared__ float rbuf[8][16][DM];  // 64 KB
  const int wv = threadIdx.x >> 5, lane = threadIdx.x & 31;
  const int half = lane >> 4, nlo = lane & 15;
  const int row0 = (blockIdx.x * 8 + wv) * 16;
  const bool active = row0 < N_DST_C;

  if (active) {
    const int arow = min(row0 + nlo, N_DST_C - 1);
    const float* r0 = agg + (size_t)arow * DM;
    const float* r1 = target_h + (size_t)arow * DM;
    v8f acc[8] = {};
    rows2_gemm<8>(r0, r1, WoB, lane, acc);
#pragma unroll
    for (int nt = 0; nt < 8; ++nt) {
      const float bias = bo[nt * 16 + nlo];
#pragma unroll
      for (int r = 0; r < 8; ++r)
        rbuf[wv][8 * half + r][nt * 16 + nlo] = fmaxf(acc[nt][r] + bias, 0.0f);
    }
  }
  __syncthreads();

  if (active && lane < 16) {
    const int row = row0 + lane;
    if (row < N_DST_C) {
      float s = 0.0f;
      for (int j = 0; j < DM; ++j) s += rbuf[wv][lane][j];
      const float mu = s * (1.0f / DM);
      float v = 0.0f;
      for (int j = 0; j < DM; ++j) {
        const float dd = rbuf[wv][lane][j] - mu;
        v = fmaf(dd, dd, v);
      }
      const float inv = rsqrtf(v * (1.0f / DM) + LN_EPS);
      for (int j = 0; j < DM; ++j)
        out[(size_t)row * DM + j] =
            fmaf(ln_g[j], (rbuf[wv][lane][j] - mu) * inv, ln_b[j]);
    }
  }
}

// ------------------------------------------------------------------ launcher
extern "C" void kernel_launch(void* const* d_in, const int* in_sizes, int n_in,
                              void* d_out, int out_size, void* d_ws, size_t ws_size,
                              hipStream_t stream) {
  const float* target_h   = (const float*)d_in[0];
  const float* source_h   = (const float*)d_in[1];
  const float* edge_feats = (const float*)d_in[2];
  const float* delta_time = (const float*)d_in[3];
  const int*   edge_dst   = (const int*)d_in[4];
  const float* tw         = (const float*)d_in[5];
  const float* tb         = (const float*)d_in[6];
  const float* Wq         = (const float*)d_in[7];
  const float* bq         = (const float*)d_in[8];
  const float* Wk         = (const float*)d_in[9];
  const float* bk         = (const float*)d_in[10];
  const float* Wv         = (const float*)d_in[11];
  const float* bv         = (const float*)d_in[12];
  const float* Wo         = (const float*)d_in[13];
  const float* bo         = (const float*)d_in[14];
  const float* ln_g       = (const float*)d_in[15];
  const float* ln_b       = (const float*)d_in[16];
  float* out = (float*)d_out;

  // Workspace layout (fp32 elements unless noted)
  float* ws = (float*)d_ws;
  float*    Qn       = ws;                         // 25000*128
  float*    scorebuf = Qn + (size_t)N_DST_C * DM;  // 400000*8
  float*    exbuf    = scorebuf + (size_t)E_C * HEADS;
  unsigned* mbuf     = (unsigned*)(exbuf + (size_t)E_C * HEADS);  // 25000*8
  float*    den      = (float*)mbuf + (size_t)N_DST_C * HEADS;
  float*    agg      = den + (size_t)N_DST_C * HEADS;             // 25000*128
  float*    bq_eff   = agg + (size_t)N_DST_C * DM;                // 128
  __bf16*   WqB = (__bf16*)(bq_eff + DM);          // 128*128 bf16
  __bf16*   WkB = WqB + 128 * 128;                 // 128*384 bf16
  __bf16*   WvB = WkB + 128 * 384;                 // 128*384 bf16
  __bf16*   WoB = WvB + 128 * 384;                 // 128*256 bf16

  // Zero the [mbuf | den | agg] region (encoded -inf for mbuf is <= 0-key).
  {
    const int nz = N_DST_C * HEADS * 2 + N_DST_C * DM;
    k_zero<<<(nz + 255) / 256, 256, 0, stream>>>((float*)mbuf, nz);
  }

  // Weight prep (tiny)
  k_prep_w<<<(128 * 128 + 255) / 256, 256, 0, stream>>>(Wq, 256, WqB, 128 * 128);
  k_prep_w<<<(128 * 384 + 255) / 256, 256, 0, stream>>>(Wk, 384, WkB, 128 * 384);
  k_prep_w<<<(128 * 384 + 255) / 256, 256, 0, stream>>>(Wv, 384, WvB, 128 * 384);
  k_prep_w<<<(128 * 256 + 255) / 256, 256, 0, stream>>>(Wo, 256, WoB, 128 * 256);
  k_prep_bq<<<1, 128, 0, stream>>>(Wq, bq, tb, bq_eff);

  const int node_blocks = (N_DST_C + 127) / 128;   // 196
  const int edge_blocks = E_C / 128;               // 3125

  k_qn<<<node_blocks, 256, 0, stream>>>(target_h, WqB, bq_eff, Qn);
  k_edge_score<<<edge_blocks, 256, 0, stream>>>(source_h, edge_feats, delta_time,
                                                edge_dst, tw, tb, bk, WkB, Qn,
                                                scorebuf, mbuf);
  k_soft<<<(E_C * HEADS + 255) / 256, 256, 0, stream>>>(scorebuf, edge_dst, mbuf,
                                                        exbuf, den);
  k_edge_agg<<<edge_blocks, 256, 0, stream>>>(source_h, edge_feats, delta_time,
                                              edge_dst, tw, tb, bv, WvB, exbuf,
                                              den, agg);
  k_out<<<node_blocks, 256, 0, stream>>>(agg, target_h, WoB, bo, ln_g, ln_b, out);
}
